// set2setReadout_90615220011133
// MI455X (gfx1250) — compile-verified, matrix operand order checked by reading, not run on hardware
//
#include <hip/hip_runtime.h>
#include <math.h>

#define HDIM 128
#define GH   512   // 4*H
#define NIT  3

typedef __attribute__((ext_vector_type(16))) _Float16 v16h;
typedef __attribute__((ext_vector_type(8)))  float    v8f;

// ---------- ordered-uint encoding for float atomicMax ----------
__device__ __forceinline__ unsigned f2ord(float f) {
  unsigned u = __float_as_uint(f);
  return (u & 0x80000000u) ? ~u : (u | 0x80000000u);
}
__device__ __forceinline__ float ord2f(unsigned e) {
  unsigned u = (e & 0x80000000u) ? (e ^ 0x80000000u) : ~e;
  return __uint_as_float(u);
}

__device__ __forceinline__ float sigm(float v) { return 1.0f / (1.0f + __expf(-v)); }

// ---------- WMMA fragment loaders (16-bit A 16x32 layout per ISA 7.12.2;
// B assumed mirrored: lane -> N column, K packed like A) ----------
__device__ __forceinline__ v16h load_a_f32(const float* __restrict__ src, int ld,
                                           int row0, int kb, int g, int hl) {
  const float* rp = src + (size_t)(row0 + hl) * ld + kb;
  v16h a;
#pragma unroll
  for (int j = 0; j < 16; ++j) {
    int k = ((j >> 3) << 4) | (g << 3) | (j & 7);
    a[j] = (_Float16)rp[k];
  }
  return a;
}
__device__ __forceinline__ v16h load_b_h(const _Float16* __restrict__ W, int ld,
                                         int colbase, int kb, int g, int hl) {
  const _Float16* rp = W + (size_t)(colbase + hl) * ld + kb;
  v16h b;
#pragma unroll
  for (int j = 0; j < 16; ++j) {
    int k = ((j >> 3) << 4) | (g << 3) | (j & 7);
    b[j] = rp[k];
  }
  return b;
}

// ---------- fused LSTM cell: g = inp@Wih^T + hprev@Whh^T + b; gate math in-register ----------
// grid: (B/16, H/16/4), block: 128 (4 waves, one 16-col tile each)
__global__ __launch_bounds__(128)
void lstm_cell_wmma(const float* __restrict__ inp, int kin,
                    const _Float16* __restrict__ Wih, const _Float16* __restrict__ Whh,
                    const float* __restrict__ b_ih, const float* __restrict__ b_hh,
                    const float* __restrict__ hprev, const float* __restrict__ cprev,
                    float* __restrict__ hout, float* __restrict__ cout) {
  const int lane = threadIdx.x & 31;
  const int wv   = threadIdx.x >> 5;
  const int row0 = blockIdx.x * 16;
  const int col0 = (blockIdx.y * 4 + wv) * 16;   // hidden-unit column tile
  const int g    = lane >> 4;
  const int hl   = lane & 15;

  v8f acc[4] = {};   // i, f, g, o gate pre-activations for this 16x16 tile

  for (int kb = 0; kb < kin; kb += 32) {
    v16h a = load_a_f32(inp, kin, row0, kb, g, hl);
#pragma unroll
    for (int gt = 0; gt < 4; ++gt) {
      v16h b = load_b_h(Wih, kin, gt * HDIM + col0, kb, g, hl);
      acc[gt] = __builtin_amdgcn_wmma_f32_16x16x32_f16(
          false, a, false, b, (short)0, acc[gt], false, false);
    }
  }
  for (int kb = 0; kb < HDIM; kb += 32) {
    v16h a = load_a_f32(hprev, HDIM, row0, kb, g, hl);
#pragma unroll
    for (int gt = 0; gt < 4; ++gt) {
      v16h b = load_b_h(Whh, HDIM, gt * HDIM + col0, kb, g, hl);
      acc[gt] = __builtin_amdgcn_wmma_f32_16x16x32_f16(
          false, a, false, b, (short)0, acc[gt], false, false);
    }
  }

  // C/D layout: VGPR r -> M = r + 8*(lane>>4), N = lane&15
  const int n  = col0 + hl;
  const float bi = b_ih[0 * HDIM + n] + b_hh[0 * HDIM + n];
  const float bf = b_ih[1 * HDIM + n] + b_hh[1 * HDIM + n];
  const float bg = b_ih[2 * HDIM + n] + b_hh[2 * HDIM + n];
  const float bo = b_ih[3 * HDIM + n] + b_hh[3 * HDIM + n];
#pragma unroll
  for (int r = 0; r < 8; ++r) {
    const int m = row0 + r + 8 * g;
    const size_t idx = (size_t)m * HDIM + n;
    const float cold = cprev[idx];
    const float iv = sigm(acc[0][r] + bi);
    const float fv = sigm(acc[1][r] + bf);
    const float gv = tanhf(acc[2][r] + bg);
    const float ov = sigm(acc[3][r] + bo);
    const float cn = fv * cold + iv * gv;
    cout[idx] = cn;
    hout[idx] = ov * tanhf(cn);
  }
}

// ---------- attention: wave per node, lane holds float4 (32*4 = 128 = H) ----------
__global__ __launch_bounds__(256)
void attn_dot(const float* __restrict__ x, const float* __restrict__ qv,
              const long long* __restrict__ ptr, float* __restrict__ e,
              unsigned* __restrict__ emax, int n) {
  const int node = (int)((blockIdx.x * (unsigned)blockDim.x + threadIdx.x) >> 5);
  const int lane = threadIdx.x & 31;
  if (node >= n) return;
  const int b = (int)ptr[node];
  const float4 xa = ((const float4*)(x + (size_t)node * HDIM))[lane];
  const float4 qa = ((const float4*)(qv + (size_t)b * HDIM))[lane];
  float s = xa.x * qa.x + xa.y * qa.y + xa.z * qa.z + xa.w * qa.w;
#pragma unroll
  for (int off = 16; off > 0; off >>= 1) s += __shfl_down(s, off, 32);
  if (lane == 0) {
    e[node] = s;
    atomicMax(&emax[b], f2ord(s));
  }
}

__global__ __launch_bounds__(256)
void attn_exp(const float* __restrict__ e, const long long* __restrict__ ptr,
              const unsigned* __restrict__ emax, float* __restrict__ ee,
              float* __restrict__ denom, int n) {
  const int i = blockIdx.x * blockDim.x + threadIdx.x;
  if (i >= n) return;
  const int b = (int)ptr[i];
  float m = ord2f(emax[b]);
  if (!isfinite(m)) m = 0.0f;
  const float v = __expf(e[i] - m);
  ee[i] = v;
  atomicAdd(&denom[b], v);
}

__global__ __launch_bounds__(256)
void attn_pool(const float* __restrict__ x, const float* __restrict__ ee,
               const float* __restrict__ denom, const long long* __restrict__ ptr,
               float* __restrict__ r, int n) {
  const int node = (int)((blockIdx.x * (unsigned)blockDim.x + threadIdx.x) >> 5);
  const int lane = threadIdx.x & 31;
  if (node >= n) return;
  const int b = (int)ptr[node];
  const float alpha = ee[node] / fmaxf(denom[b], 1e-16f);
  const float4 xa = ((const float4*)(x + (size_t)node * HDIM))[lane];
  float* rb = r + (size_t)b * HDIM + lane * 4;
  atomicAdd(rb + 0, alpha * xa.x);
  atomicAdd(rb + 1, alpha * xa.y);
  atomicAdd(rb + 2, alpha * xa.z);
  atomicAdd(rb + 3, alpha * xa.w);
}

// ---------- misc small kernels ----------
__global__ void zero_f32(float* p, int n) {
  int i = blockIdx.x * blockDim.x + threadIdx.x;
  if (i < n) p[i] = 0.0f;
}
__global__ void w2h(const float* __restrict__ src, _Float16* __restrict__ dst, int n) {
  int i = blockIdx.x * blockDim.x + threadIdx.x;
  if (i < n) dst[i] = (_Float16)src[i];
}
__global__ void seg_init(unsigned* emax, float* denom, float* r, int B) {
  int i = blockIdx.x * blockDim.x + threadIdx.x;
  if (i < B) { emax[i] = 0u; denom[i] = 0.0f; }
  if (i < B * HDIM) r[i] = 0.0f;
}
__global__ void concat_q(const float* __restrict__ h1, const float* __restrict__ r,
                         float* __restrict__ q, int B) {
  int i = blockIdx.x * blockDim.x + threadIdx.x;
  if (i >= B * 2 * HDIM) return;
  const int b = i / (2 * HDIM);
  const int j = i - b * 2 * HDIM;
  q[i] = (j < HDIM) ? h1[(size_t)b * HDIM + j] : r[(size_t)b * HDIM + (j - HDIM)];
}

extern "C" void kernel_launch(void* const* d_in, const int* in_sizes, int n_in,
                              void* d_out, int out_size, void* d_ws, size_t ws_size,
                              hipStream_t stream) {
  const float*     x     = (const float*)d_in[0];
  const float*     W_ih0 = (const float*)d_in[1];
  const float*     W_hh0 = (const float*)d_in[2];
  const float*     b_ih0 = (const float*)d_in[3];
  const float*     b_hh0 = (const float*)d_in[4];
  const float*     W_ih1 = (const float*)d_in[5];
  const float*     W_hh1 = (const float*)d_in[6];
  const float*     b_ih1 = (const float*)d_in[7];
  const float*     b_hh1 = (const float*)d_in[8];
  const long long* ptr   = (const long long*)d_in[9];

  const int N = in_sizes[0] / HDIM;
  const int B = out_size / (2 * HDIM);   // 1024

  // ---- workspace layout (256B aligned slices) ----
  char* w = (char*)d_ws;
  auto take = [&](size_t bytes) { char* p = w; w += (bytes + 255) & ~(size_t)255; return p; };
  _Float16* Wih0h = (_Float16*)take((size_t)GH * 2 * HDIM * 2);
  _Float16* Whh0h = (_Float16*)take((size_t)GH * HDIM * 2);
  _Float16* Wih1h = (_Float16*)take((size_t)GH * HDIM * 2);
  _Float16* Whh1h = (_Float16*)take((size_t)GH * HDIM * 2);
  float* q    = (float*)take((size_t)B * 2 * HDIM * 4);
  float* h0a  = (float*)take((size_t)B * HDIM * 4);
  float* h0b  = (float*)take((size_t)B * HDIM * 4);
  float* c0a  = (float*)take((size_t)B * HDIM * 4);
  float* c0b  = (float*)take((size_t)B * HDIM * 4);
  float* h1a  = (float*)take((size_t)B * HDIM * 4);
  float* h1b  = (float*)take((size_t)B * HDIM * 4);
  float* c1a  = (float*)take((size_t)B * HDIM * 4);
  float* c1b  = (float*)take((size_t)B * HDIM * 4);
  float* e    = (float*)take((size_t)N * 4);
  float* ee   = (float*)take((size_t)N * 4);
  unsigned* emax = (unsigned*)take((size_t)B * 4);
  float* denom   = (float*)take((size_t)B * 4);
  float* r       = (float*)take((size_t)B * HDIM * 4);
  (void)ws_size; (void)n_in;

  // ---- one-time (per call) prep: f16 weights + zero initial state ----
  const int bs = 256;
  w2h<<<(GH * 2 * HDIM + bs - 1) / bs, bs, 0, stream>>>(W_ih0, Wih0h, GH * 2 * HDIM);
  w2h<<<(GH * HDIM + bs - 1) / bs, bs, 0, stream>>>(W_hh0, Whh0h, GH * HDIM);
  w2h<<<(GH * HDIM + bs - 1) / bs, bs, 0, stream>>>(W_ih1, Wih1h, GH * HDIM);
  w2h<<<(GH * HDIM + bs - 1) / bs, bs, 0, stream>>>(W_hh1, Whh1h, GH * HDIM);
  zero_f32<<<(B * 2 * HDIM + bs - 1) / bs, bs, 0, stream>>>(q, B * 2 * HDIM);
  zero_f32<<<(B * HDIM + bs - 1) / bs, bs, 0, stream>>>(h0a, B * HDIM);
  zero_f32<<<(B * HDIM + bs - 1) / bs, bs, 0, stream>>>(c0a, B * HDIM);
  zero_f32<<<(B * HDIM + bs - 1) / bs, bs, 0, stream>>>(h1a, B * HDIM);
  zero_f32<<<(B * HDIM + bs - 1) / bs, bs, 0, stream>>>(c1a, B * HDIM);

  float *h0p = h0a, *c0p = c0a, *h0n = h0b, *c0n = c0b;
  float *h1p = h1a, *c1p = c1a, *h1n = h1b, *c1n = c1b;

  const dim3 gLstm(B / 16, HDIM / 16 / 4);       // 64 x 2 blocks, 4 waves each
  const int nodeBlocks = (N * 32 + bs - 1) / bs; // wave per node

  for (int it = 0; it < NIT; ++it) {
    lstm_cell_wmma<<<gLstm, 128, 0, stream>>>(q, 2 * HDIM, Wih0h, Whh0h,
                                              b_ih0, b_hh0, h0p, c0p, h0n, c0n);
    lstm_cell_wmma<<<gLstm, 128, 0, stream>>>(h0n, HDIM, Wih1h, Whh1h,
                                              b_ih1, b_hh1, h1p, c1p, h1n, c1n);

    seg_init<<<(B * HDIM + bs - 1) / bs, bs, 0, stream>>>(emax, denom, r, B);
    attn_dot<<<nodeBlocks, bs, 0, stream>>>(x, h1n, ptr, e, emax, N);
    attn_exp<<<(N + bs - 1) / bs, bs, 0, stream>>>(e, ptr, emax, ee, denom, N);
    attn_pool<<<nodeBlocks, bs, 0, stream>>>(x, ee, denom, ptr, r, N);

    float* qdst = (it == NIT - 1) ? (float*)d_out : q;
    concat_q<<<(B * 2 * HDIM + bs - 1) / bs, bs, 0, stream>>>(h1n, r, qdst, B);

    float* t;
    t = h0p; h0p = h0n; h0n = t;  t = c0p; c0p = c0n; c0n = t;
    t = h1p; h1p = h1n; h1n = t;  t = c1p; c1p = c1n; c1n = t;
  }
}
